// AGCRN_24232205484697
// MI455X (gfx1250) — matrix operand order repaired and checked by reading, not compiled, not run on hardware
//
#include <hip/hip_runtime.h>
#include <hip/hip_bf16.h>
#include <cstdint>

// ---------------------------------------------------------------------------
// AGCRN forward for MI455X (gfx1250): bf16 WMMA (16x16x32) + f32 accumulate.
// All matmuls (graph mixing, per-node adaptive GEMMs, output head) lower to
// v_wmma_f32_16x16x32_bf16. Layouts make every WMMA fragment two contiguous
// 16B loads per lane (CDNA5 A/B VGPR striping). GEMM kernels are templated so
// K-loops fully unroll and tile bookkeeping is compile-time; waves block over
// N (2 output tiles / both supports) to reuse each A fragment across 2 WMMAs.
// ---------------------------------------------------------------------------

typedef __bf16 bf16;
typedef bf16  bf16x8 __attribute__((ext_vector_type(8)));
typedef bf16  v16bf  __attribute__((ext_vector_type(16)));
typedef float v8f    __attribute__((ext_vector_type(8)));

// problem constants
static constexpr int N_  = 134;   // nodes
static constexpr int B_  = 32;    // batch
static constexpr int T_  = 288;   // timesteps
static constexpr int V_  = 12;    // raw vars
static constexpr int H_  = 64;    // hidden
static constexpr int E_  = 10;    // embedding
static constexpr int OW_ = 288;   // output window
static constexpr int NP_ = 144;   // nodes padded to 16 (output/N dim of mixing)
static constexpr int MP_ = 160;   // nodes padded to 32 (K dim of mixing)

#define DEV static __device__ __forceinline__

// A fragment: 16x32 bf16 tile, row-major source with stride ld.
// Lane l (l<16): row=l, K={0..7}u{16..23}; lane l+16: row=l, K={8..15}u{24..31}.
DEV v16bf load_frag_a(const bf16* base, int ld) {
  int lane = threadIdx.x & 31;
  int row  = lane & 15;
  int half = lane >> 4;
  const bf16* p = base + (size_t)row * ld + half * 8;
  bf16x8 lo = *reinterpret_cast<const bf16x8*>(p);        // K: half*8 + 0..7
  bf16x8 hi = *reinterpret_cast<const bf16x8*>(p + 16);   // K: 16 + half*8 + 0..7
  v16bf r;
#pragma unroll
  for (int j = 0; j < 8; ++j) { r[j] = lo[j]; r[j + 8] = hi[j]; }
  return r;
}

// B fragment: 32x16 bf16 tile stored column-major (each output column is
// contiguous over K, stride ld between columns).
// Lanes 0-15: col=lane, K={0..7}u{16..23}; lanes 16-31: col=lane-16, K={8..15}u{24..31}.
DEV v16bf load_frag_b(const bf16* base, int ld) {
  int lane = threadIdx.x & 31;
  int col  = lane & 15;
  int half = lane >> 4;
  const bf16* p = base + (size_t)col * ld + half * 8;
  bf16x8 lo = *reinterpret_cast<const bf16x8*>(p);
  bf16x8 hi = *reinterpret_cast<const bf16x8*>(p + 16);
  v16bf r;
#pragma unroll
  for (int j = 0; j < 8; ++j) { r[j] = lo[j]; r[j + 8] = hi[j]; }
  return r;
}

DEV v8f wmma_bf16(v16bf a, v16bf b, v8f c) {
  return __builtin_amdgcn_wmma_f32_16x16x32_bf16(
      /*neg_a=*/false, a, /*neg_b=*/false, b,
      /*c_mod=*/(short)0, c, /*reuse_a=*/false, /*reuse_b=*/false);
}

// ---------------------------------------------------------------------------
// 1) Preprocess: normalize + diff channel -> xseq0[t][n][b][c] bf16, c-pad 16.
// ---------------------------------------------------------------------------
__global__ void k_pre(const float* __restrict__ bx, const float* __restrict__ mean,
                      const float* __restrict__ scale, bf16* __restrict__ xseq0) {
  long idx = (long)blockIdx.x * blockDim.x + threadIdx.x;
  long total = (long)T_ * N_ * B_ * 16;
  if (idx >= total) return;
  int c = (int)(idx & 15); long r = idx >> 4;
  int b = (int)(r % B_); r /= B_;
  int n = (int)(r % N_);
  int t = (int)(r / N_);
  float v = 0.f;
  if (c < 10) {
    float m = mean[n * 10 + c], s = scale[n * 10 + c];
    v = (bx[(((long)b * N_ + n) * T_ + t) * V_ + (c + 2)] - m) / s;
  } else if (c == 10 && t > 0) {
    float m = mean[n * 10 + 9], s = scale[n * 10 + 9];
    float a = (bx[(((long)b * N_ + n) * T_ + t) * V_ + 11] - m) / s;
    float p = (bx[(((long)b * N_ + n) * T_ + (t - 1)) * V_ + 11] - m) / s;
    v = a - p;
  }
  xseq0[idx] = (bf16)v;
}

// ---------------------------------------------------------------------------
// 2) Supports: pack lap (padded) + apt = softmax(relu(emb emb^T)) into
//    S2[2][NP_][MP_] bf16, row-major (row n over m).
// ---------------------------------------------------------------------------
__global__ void k_supports(const float* __restrict__ emb, const float* __restrict__ lap,
                           bf16* __restrict__ S2) {
  __shared__ float sc[MP_];
  __shared__ float red[256];
  int n = blockIdx.x;
  int tid = threadIdx.x;
  bf16* lapO = S2;
  bf16* aptO = S2 + (size_t)NP_ * MP_;
  if (n >= N_) {
    for (int m = tid; m < MP_; m += blockDim.x) { lapO[n*MP_+m] = (bf16)0.f; aptO[n*MP_+m] = (bf16)0.f; }
    return;
  }
  for (int m = tid; m < MP_; m += blockDim.x)
    lapO[n*MP_+m] = (bf16)((m < N_) ? lap[n*N_+m] : 0.f);
  for (int m = tid; m < MP_; m += blockDim.x) {
    float d = 0.f;
    if (m < N_) {
      for (int e = 0; e < E_; ++e) d += emb[n*E_+e] * emb[m*E_+e];
      d = d > 0.f ? d : 0.f;
    }
    sc[m] = (m < N_) ? d : -3.0e38f;
  }
  __syncthreads();
  float mx = -3.0e38f;
  for (int m = tid; m < MP_; m += blockDim.x) mx = fmaxf(mx, sc[m]);
  red[tid] = mx; __syncthreads();
  for (int s = 128; s > 0; s >>= 1) { if (tid < s) red[tid] = fmaxf(red[tid], red[tid+s]); __syncthreads(); }
  mx = red[0]; __syncthreads();
  float sum = 0.f;
  for (int m = tid; m < MP_; m += blockDim.x) {
    float e2 = (m < N_) ? __expf(sc[m] - mx) : 0.f;
    sc[m] = e2; sum += e2;
  }
  __syncthreads();
  red[tid] = sum; __syncthreads();
  for (int s = 128; s > 0; s >>= 1) { if (tid < s) red[tid] += red[tid+s]; __syncthreads(); }
  float inv = 1.f / red[0];
  __syncthreads();
  for (int m = tid; m < MP_; m += blockDim.x) aptO[n*MP_+m] = (bf16)(sc[m] * inv);
}

// ---------------------------------------------------------------------------
// 3) Node-adaptive weights: Wn[n][o][k*Cp+i] = sum_d emb[n][d]*W[d][k][i][o]
//    (zero-padded i>=Cin) stored so B fragments are contiguous. bias likewise.
// ---------------------------------------------------------------------------
__global__ void k_mkw(const float* __restrict__ emb, const float* __restrict__ W,
                      const float* __restrict__ bsrc, bf16* __restrict__ Wn,
                      float* __restrict__ bn, int K, int Cin, int Cp, int Cout) {
  long idx = (long)blockIdx.x * blockDim.x + threadIdx.x;
  int kcT = K * Cp;
  long total = (long)N_ * Cout * kcT;
  if (idx >= total) return;
  int kc = (int)(idx % kcT); long r = idx / kcT;
  int o = (int)(r % Cout);
  int n = (int)(r / Cout);
  int k = kc / Cp, i = kc % Cp;
  float v = 0.f;
  if (i < Cin) {
    for (int e = 0; e < E_; ++e)
      v += emb[n*E_+e] * W[(((long)e * K + k) * Cin + i) * Cout + o];
  }
  Wn[idx] = (bf16)v;
  if (kc == 0) {
    float bb = 0.f;
    for (int e = 0; e < E_; ++e) bb += emb[n*E_+e] * bsrc[e*Cout+o];
    bn[n*Cout+o] = bb;
  }
}

__global__ void k_cvt(const float* __restrict__ src, bf16* __restrict__ dst, int n) {
  int i = blockIdx.x * blockDim.x + threadIdx.x;
  if (i < n) dst[i] = (bf16)src[i];
}

__global__ void k_zero(float* __restrict__ p, int n) {
  int i = blockIdx.x * blockDim.x + threadIdx.x;
  if (i < n) p[i] = 0.f;
}

// ---------------------------------------------------------------------------
// 4) Build recurrent input: InpT[(b*Cp+c)][m] (bf16, m-padded to MP_) and the
//    identity-support slice XG[n][b][0*Cp+c]. useZ=1 -> candidate path (z*h).
// ---------------------------------------------------------------------------
__global__ void k_build(const bf16* __restrict__ xin, const float* __restrict__ hbuf,
                        const float* __restrict__ zr, int useZ,
                        bf16* __restrict__ inpT, bf16* __restrict__ xg,
                        int Cx, int Cxpad, int Cp, int KCpad) {
  int idx = blockIdx.x * blockDim.x + threadIdx.x;
  int total = B_ * Cp * MP_;
  if (idx >= total) return;
  int m  = idx % MP_;
  int bc = idx / MP_;
  int c = bc % Cp, b = bc / Cp;
  float v = 0.f;
  if (m < N_) {
    if (c < Cx) {
      v = (float)xin[((size_t)m * B_ + b) * Cxpad + c];
    } else if (c < Cx + H_) {
      int hc = c - Cx;
      float hv = hbuf[((size_t)m * B_ + b) * H_ + hc];
      if (useZ) hv *= zr[((size_t)m * B_ + b) * (2 * H_) + hc];   // z = first half
      v = hv;
    }
  }
  inpT[idx] = (bf16)v;
  if (m < N_) xg[((size_t)m * B_ + b) * KCpad + c] = (bf16)v;
}

// ---------------------------------------------------------------------------
// 5) Graph mixing GEMM (WMMA): one wave task computes BOTH supports for a
//    16x16 tile (A fragment reused across 2 WMMAs per K-step):
//    Out_s[bc][n] = sum_m InpT[bc][m] * S[s][n][m] -> XG[n][b][(s+1)*Cp + c]
// ---------------------------------------------------------------------------
template <int Cp>
__global__ void __launch_bounds__(256)
k_mix(const bf16* __restrict__ inpT, const bf16* __restrict__ S2,
      bf16* __restrict__ xg) {
  constexpr int KCpad = 3 * Cp;
  constexpr int Mtiles = (B_ * Cp) / 16;
  constexpr int Ntiles = NP_ / 16;            // 9
  int task = blockIdx.x * (blockDim.x >> 5) + (threadIdx.x >> 5);
  if (task >= Mtiles * Ntiles) return;
  int nt = task % Ntiles;
  int mt = task / Ntiles;
  const bf16* A0 = inpT + (size_t)mt * 16 * MP_;
  const bf16* B0 = S2 + (size_t)nt * 16 * MP_;          // lap
  const bf16* B1 = B0 + (size_t)NP_ * MP_;              // apt
  v8f acc0 = {}, acc1 = {};
#pragma unroll
  for (int ks = 0; ks < MP_ / 32; ++ks) {     // 5 K-steps over padded nodes
    v16bf a  = load_frag_a(A0 + ks * 32, MP_);
    v16bf b0 = load_frag_b(B0 + ks * 32, MP_);
    v16bf b1 = load_frag_b(B1 + ks * 32, MP_);
    acc0 = wmma_bf16(a, b0, acc0);
    acc1 = wmma_bf16(a, b1, acc1);
  }
  int lane = threadIdx.x & 31;
  int col = lane & 15, half = lane >> 4;
  int n = nt * 16 + col;
  if (n >= N_) return;
  int rowbase = mt * 16 + half * 8;           // Cp multiple of 16 -> same b
  int b = rowbase / Cp, c0 = rowbase % Cp;    // compile-time Cp
  bf16* d0 = xg + ((size_t)n * B_ + b) * KCpad + 1 * Cp + c0;
  bf16* d1 = xg + ((size_t)n * B_ + b) * KCpad + 2 * Cp + c0;
#pragma unroll
  for (int j = 0; j < 8; ++j) { d0[j] = (bf16)acc0[j]; d1[j] = (bf16)acc1[j]; }
}

// ---------------------------------------------------------------------------
// 6) Per-node WMMA GEMM: (32 x KCpad) @ (KCpad x Cout) + bias, fused GRU math.
//    MODE 0: gate -> zr = sigmoid(.)  MODE 1: update -> h = r*h + (1-r)*tanh(.)
//    NBLK output tiles per wave share each A fragment. Compile-time K-loop.
// ---------------------------------------------------------------------------
template <int KCpad, int Cout, int MODE, int NBLK>
__global__ void __launch_bounds__(256)
k_gemm_node(const bf16* __restrict__ xg, const bf16* __restrict__ wn,
            const float* __restrict__ bn, float* __restrict__ zrOut,
            const float* __restrict__ zrIn, float* __restrict__ hbuf,
            bf16* __restrict__ seqOut) {
  constexpr int groups = (Cout / 16) / NBLK;
  constexpr int tasks = (B_ / 16) * groups;   // 8 for all instantiations
  int n = blockIdx.x;
  int wv = threadIdx.x >> 5;
  const bf16* A0 = xg + (size_t)n * B_ * KCpad;
  const bf16* W0 = wn + (size_t)n * Cout * KCpad;
  for (int task = wv; task < tasks; task += 8) {
    int og = task % groups, mt = task / groups;
    const bf16* Ab = A0 + (size_t)mt * 16 * KCpad;
    const bf16* Bb = W0 + (size_t)og * NBLK * 16 * KCpad;
    __builtin_prefetch(Bb, 0, 2);             // weight stream (L2 resident)
    v8f acc[NBLK];
    v8f zacc = {};
#pragma unroll
    for (int q = 0; q < NBLK; ++q) acc[q] = zacc;
#pragma unroll
    for (int ks = 0; ks < KCpad / 32; ++ks) {
      v16bf a = load_frag_a(Ab + ks * 32, KCpad);
#pragma unroll
      for (int q = 0; q < NBLK; ++q) {
        v16bf b = load_frag_b(Bb + (size_t)q * 16 * KCpad + ks * 32, KCpad);
        acc[q] = wmma_bf16(a, b, acc[q]);
      }
    }
    int lane = threadIdx.x & 31;
    int col = lane & 15, half = lane >> 4;
    int b0 = mt * 16 + half * 8;
#pragma unroll
    for (int q = 0; q < NBLK; ++q) {
      int o = (og * NBLK + q) * 16 + col;
      float bias = bn[n * Cout + o];
#pragma unroll
      for (int j = 0; j < 8; ++j) {
        int b = b0 + j;
        float v = acc[q][j] + bias;
        if (MODE == 0) {
          zrOut[((size_t)n * B_ + b) * Cout + o] = 1.f / (1.f + __expf(-v));
        } else {
          float hc = tanhf(v);
          size_t hidx = ((size_t)n * B_ + b) * H_ + o;
          float rg = zrIn[((size_t)n * B_ + b) * (2 * H_) + H_ + o]; // r = 2nd half
          float hn = rg * hbuf[hidx] + (1.f - rg) * hc;
          hbuf[hidx] = hn;
          seqOut[hidx] = (bf16)hn;
        }
      }
    }
  }
}

// ---------------------------------------------------------------------------
// 7) Output head: out[b][n][ow] = h_last[n][b][:] @ convW[ow][:] + convb[ow]
//    2-wide N blocking, compile-time 2-step K loop.
// ---------------------------------------------------------------------------
__global__ void __launch_bounds__(256)
k_head(const bf16* __restrict__ hlast, const bf16* __restrict__ wcv,
       const float* __restrict__ bcv, float* __restrict__ out) {
  constexpr int groups = (OW_ / 16) / 2;      // 9
  constexpr int tasks = (B_ / 16) * groups;   // 18
  int n = blockIdx.x;
  int wv = threadIdx.x >> 5;
  const bf16* A0 = hlast + (size_t)n * B_ * H_;
  for (int task = wv; task < tasks; task += 8) {
    int og = task % groups, mt = task / groups;
    const bf16* Ab = A0 + (size_t)mt * 16 * H_;
    v8f acc0 = {}, acc1 = {};
#pragma unroll
    for (int ks = 0; ks < H_ / 32; ++ks) {    // 2 K-steps
      v16bf a  = load_frag_a(Ab + ks * 32, H_);
      v16bf b0 = load_frag_b(wcv + (size_t)(og * 2 + 0) * 16 * H_ + ks * 32, H_);
      v16bf b1 = load_frag_b(wcv + (size_t)(og * 2 + 1) * 16 * H_ + ks * 32, H_);
      acc0 = wmma_bf16(a, b0, acc0);
      acc1 = wmma_bf16(a, b1, acc1);
    }
    int lane = threadIdx.x & 31;
    int col = lane & 15, half = lane >> 4;
    int b0r = mt * 16 + half * 8;
#pragma unroll
    for (int q = 0; q < 2; ++q) {
      int ow = (og * 2 + q) * 16 + col;
      float bias = bcv[ow];
#pragma unroll
      for (int j = 0; j < 8; ++j) {
        float v = (q ? acc1[j] : acc0[j]) + bias;
        out[((size_t)(b0r + j) * N_ + n) * OW_ + ow] = v;
      }
    }
  }
}

// ---------------------------------------------------------------------------
extern "C" void kernel_launch(void* const* d_in, const int* in_sizes, int n_in,
                              void* d_out, int out_size, void* d_ws, size_t ws_size,
                              hipStream_t stream) {
  (void)in_sizes; (void)n_in; (void)out_size; (void)ws_size;
  const float* bx    = (const float*)d_in[0];
  const float* dmean = (const float*)d_in[2];
  const float* dscal = (const float*)d_in[3];
  const float* lap   = (const float*)d_in[4];
  const float* emb   = (const float*)d_in[5];
  const float* l0gW  = (const float*)d_in[6];
  const float* l0gb  = (const float*)d_in[7];
  const float* l0uW  = (const float*)d_in[8];
  const float* l0ub  = (const float*)d_in[9];
  const float* l1gW  = (const float*)d_in[10];
  const float* l1gb  = (const float*)d_in[11];
  const float* l1uW  = (const float*)d_in[12];
  const float* l1ub  = (const float*)d_in[13];
  const float* convW = (const float*)d_in[14];
  const float* convb = (const float*)d_in[15];
  float* out = (float*)d_out;

  char* ws = (char*)d_ws;
  size_t off = 0;
  auto carve = [&](size_t bytes) -> void* {
    void* p = ws + off;
    off += (bytes + 255) & ~(size_t)255;
    return p;
  };

  bf16*  xseq0 = (bf16*)carve((size_t)T_ * N_ * B_ * 16 * sizeof(bf16));
  bf16*  xseq1 = (bf16*)carve((size_t)T_ * N_ * B_ * H_ * sizeof(bf16));
  bf16*  wnl0g = (bf16*)carve((size_t)N_ * 128 * 288 * sizeof(bf16));
  bf16*  wnl0u = (bf16*)carve((size_t)N_ * 64  * 288 * sizeof(bf16));
  bf16*  wnl1g = (bf16*)carve((size_t)N_ * 128 * 384 * sizeof(bf16));
  bf16*  wnl1u = (bf16*)carve((size_t)N_ * 64  * 384 * sizeof(bf16));
  float* bnl0g = (float*)carve((size_t)N_ * 128 * 4);
  float* bnl0u = (float*)carve((size_t)N_ * 64  * 4);
  float* bnl1g = (float*)carve((size_t)N_ * 128 * 4);
  float* bnl1u = (float*)carve((size_t)N_ * 64  * 4);
  bf16*  convWb= (bf16*)carve((size_t)OW_ * H_ * sizeof(bf16));
  bf16*  S2    = (bf16*)carve((size_t)2 * NP_ * MP_ * sizeof(bf16));
  bf16*  inpT  = (bf16*)carve((size_t)B_ * 128 * MP_ * sizeof(bf16));  // max layer
  bf16*  xg    = (bf16*)carve((size_t)N_ * B_ * 384 * sizeof(bf16));   // max layer
  float* zr    = (float*)carve((size_t)N_ * B_ * 128 * 4);
  float* hbuf  = (float*)carve((size_t)N_ * B_ * H_ * 4);
  bf16*  hlast = (bf16*)carve((size_t)N_ * B_ * H_ * sizeof(bf16));

  // --- one-time setup ---
  {
    long tot = (long)T_ * N_ * B_ * 16;
    k_pre<<<(unsigned)((tot + 255) / 256), 256, 0, stream>>>(bx, dmean, dscal, xseq0);
  }
  k_supports<<<NP_, 256, 0, stream>>>(emb, lap, S2);
  {
    long t0 = (long)N_ * 128 * 288;
    k_mkw<<<(unsigned)((t0 + 255) / 256), 256, 0, stream>>>(emb, l0gW, l0gb, wnl0g, bnl0g, 3, 75, 96, 128);
    long t1 = (long)N_ * 64 * 288;
    k_mkw<<<(unsigned)((t1 + 255) / 256), 256, 0, stream>>>(emb, l0uW, l0ub, wnl0u, bnl0u, 3, 75, 96, 64);
    long t2 = (long)N_ * 128 * 384;
    k_mkw<<<(unsigned)((t2 + 255) / 256), 256, 0, stream>>>(emb, l1gW, l1gb, wnl1g, bnl1g, 3, 128, 128, 128);
    long t3 = (long)N_ * 64 * 384;
    k_mkw<<<(unsigned)((t3 + 255) / 256), 256, 0, stream>>>(emb, l1uW, l1ub, wnl1u, bnl1u, 3, 128, 128, 64);
  }
  k_cvt<<<(OW_ * H_ + 255) / 256, 256, 0, stream>>>(convW, convWb, OW_ * H_);

  // --- recurrent layers ---
  for (int L = 0; L < 2; ++L) {
    const bf16* xin = (L == 0) ? xseq0 : xseq1;
    int Cx  = (L == 0) ? 11 : 64;
    int Cxp = (L == 0) ? 16 : 64;
    int Cp  = (L == 0) ? 96 : 128;
    int KCpad = 3 * Cp;
    bf16*  wg = (L == 0) ? wnl0g : wnl1g;
    float* bg = (L == 0) ? bnl0g : bnl1g;
    bf16*  wu = (L == 0) ? wnl0u : wnl1u;
    float* bu = (L == 0) ? bnl0u : bnl1u;

    int hElems = N_ * B_ * H_;
    k_zero<<<(hElems + 255) / 256, 256, 0, stream>>>(hbuf, hElems);

    int buildTot = B_ * Cp * MP_;
    unsigned buildGrid = (unsigned)((buildTot + 255) / 256);
    int mixTasks = ((B_ * Cp) / 16) * (NP_ / 16);     // one task = both supports
    unsigned mixGrid = (unsigned)((mixTasks + 7) / 8);

    for (int t = 0; t < T_; ++t) {
      const bf16* xt = xin + (size_t)t * N_ * B_ * Cxp;
      bf16* seqOut = (L == 0) ? (xseq1 + (size_t)t * N_ * B_ * H_) : hlast;
      // gate: zr = sigmoid(gconv([x,h]))
      k_build<<<buildGrid, 256, 0, stream>>>(xt, hbuf, zr, 0, inpT, xg, Cx, Cxp, Cp, KCpad);
      if (L == 0) {
        k_mix<96><<<mixGrid, 256, 0, stream>>>(inpT, S2, xg);
        k_gemm_node<288, 128, 0, 2><<<N_, 256, 0, stream>>>(xg, wg, bg, zr, nullptr, nullptr, nullptr);
      } else {
        k_mix<128><<<mixGrid, 256, 0, stream>>>(inpT, S2, xg);
        k_gemm_node<384, 128, 0, 2><<<N_, 256, 0, stream>>>(xg, wg, bg, zr, nullptr, nullptr, nullptr);
      }
      // candidate + GRU update: h = r*h + (1-r)*tanh(gconv([x,z*h]))
      k_build<<<buildGrid, 256, 0, stream>>>(xt, hbuf, zr, 1, inpT, xg, Cx, Cxp, Cp, KCpad);
      if (L == 0) {
        k_mix<96><<<mixGrid, 256, 0, stream>>>(inpT, S2, xg);
        k_gemm_node<288, 64, 1, 1><<<N_, 256, 0, stream>>>(xg, wu, bu, nullptr, zr, hbuf, seqOut);
      } else {
        k_mix<128><<<mixGrid, 256, 0, stream>>>(inpT, S2, xg);
        k_gemm_node<384, 64, 1, 1><<<N_, 256, 0, stream>>>(xg, wu, bu, nullptr, zr, hbuf, seqOut);
      }
    }
  }

  // --- output head ---
  k_head<<<N_, 256, 0, stream>>>(hlast, convWb, convb, out);
}
